// MPGNN_33088428048939
// MI455X (gfx1250) — matrix operand order repaired and checked by reading, not compiled, not run on hardware
//
#include <hip/hip_runtime.h>
#include <hip/hip_bf16.h>
#include <math.h>

// Problem sizes (fixed by the reference)
#define Bn 64
#define Nn 1024
#define Dn 128
#define Hn 64

// ---- vector types ----
typedef __attribute__((ext_vector_type(16))) __bf16 v16bf;
typedef __attribute__((ext_vector_type(8)))  __bf16 v8bf;
typedef __attribute__((ext_vector_type(4)))  __bf16 v4bf;
typedef __attribute__((ext_vector_type(8)))  float  v8f;
typedef __attribute__((ext_vector_type(4)))  float  f4;

union V16 { v16bf v; v4bf q[4]; };
union V8  { v8bf  v; v4bf q[2]; };
union F8  { v8f   v; float f[8]; };

__device__ __forceinline__ v4bf cvt4(f4 v) {
  v4bf o; o[0]=(__bf16)v[0]; o[1]=(__bf16)v[1]; o[2]=(__bf16)v[2]; o[3]=(__bf16)v[3]; return o;
}
__device__ __forceinline__ v4bf tanh4(f4 v) {
  v4bf o; o[0]=(__bf16)tanhf(v[0]); o[1]=(__bf16)tanhf(v[1]);
  o[2]=(__bf16)tanhf(v[2]); o[3]=(__bf16)tanhf(v[3]); return o;
}

// Async global -> LDS copy, 16B per lane. dst is the wave-relative LDS byte
// offset (addr[31:0] of a generic LDS pointer), src a 64-bit global address.
// Tracked by ASYNCcnt; completion is in-order among async loads.
__device__ __forceinline__ void async_b128(unsigned lds_off, const void* gsrc) {
  asm volatile("global_load_async_to_lds_b128 %0, %1, off"
               :: "v"(lds_off), "v"(gsrc) : "memory");
}

// ---------------------------------------------------------------------------
// Kernel 1: Tt[b][d][n] = bf16(tanh(x[b][n][d]))  (transposed so kernel 2's
// B-operand chunks are contiguous). 64x64 tiles through LDS.
// grid = B * (N/64) * (D/64) = 2048 blocks, 256 threads.
// ---------------------------------------------------------------------------
__global__ void __launch_bounds__(256)
k1_tanh_transpose(const float* __restrict__ x, __bf16* __restrict__ Tt) {
  __shared__ __bf16 sh[64 * 68];          // 64 rows of 64 halfs, stride 68 (8B-aligned rows)
  const int t  = threadIdx.x;
  const int bz = blockIdx.x;
  const int b  = bz >> 5;                 // 32 tiles per batch
  const int rem = bz & 31;
  const int n0 = (rem >> 1) * 64;
  const int d0 = (rem & 1) * 64;

  // read 64(n) x 64(d) f32 tile, tanh, cast, stash row-major in LDS
  {
    const int nl  = t >> 2;
    const int dl0 = (t & 3) * 16;
    const float* xp = x + (((size_t)b * Nn + n0 + nl) * Dn + d0 + dl0);
    #pragma unroll
    for (int i = 0; i < 4; ++i) {
      f4 v = ((const f4*)xp)[i];
      *((v4bf*)&sh[nl * 68 + dl0 + 4 * i]) = tanh4(v);
    }
  }
  __syncthreads();
  // write transposed: 16 n-contiguous halfs per thread (2 x b128 stores)
  {
    const int dl  = t >> 2;
    const int nl0 = (t & 3) * 16;
    union { v8bf v[2]; __bf16 h[16]; } u;
    #pragma unroll
    for (int i = 0; i < 16; ++i) u.h[i] = sh[(nl0 + i) * 68 + dl];
    __bf16* op = Tt + (((size_t)b * Dn + d0 + dl) * Nn + n0 + nl0);
    ((v8bf*)op)[0] = u.v[0];
    ((v8bf*)op)[1] = u.v[1];
  }
}

// ---------------------------------------------------------------------------
// Kernel 2 (dominant): M[b][n][d] = bf16(tanh( L[b] @ tanh(x[b]) ))
// Block tile: 128 rows x 128 cols (full D), K chunked by 32, double-buffered.
//  - L chunk: coalesced f32 loads -> regs -> cvt bf16 -> ds_store (needs cvt)
//  - Tt chunk: GLOBAL_LOAD_ASYNC_TO_LDS_B128 (pure bf16 copy, ASYNCcnt)
// Software pipeline: issue next chunk's async + global loads before the single
// per-iteration barrier; s_wait_asynccnt 2 retires the current chunk's pair.
// grid = B * (N/128) = 512 blocks, 256 threads (8 waves).
// ---------------------------------------------------------------------------
__global__ void __launch_bounds__(256)
k2_msg_gemm(const float* __restrict__ L, const __bf16* __restrict__ Tt,
            __bf16* __restrict__ M) {
  __shared__ __bf16 Ash[2][128 * 36];     // L chunk  [row][k], stride 36 halfs
  __shared__ __bf16 Tsh[2][128 * 40];     // Tt chunk [d][k],  stride 40 halfs (16B-aligned rows)
  const int t    = threadIdx.x;
  const int b    = blockIdx.x >> 3;
  const int row0 = (blockIdx.x & 7) * 128;
  const int w    = t >> 5;
  const int lane = t & 31;
  const int ll   = lane & 15;
  const int hi   = lane >> 4;

  v8f c[8];
  const v8f zero = {0.f, 0.f, 0.f, 0.f, 0.f, 0.f, 0.f, 0.f};
  #pragma unroll
  for (int i = 0; i < 8; ++i) c[i] = zero;

  // staging assignments: 2 threads per row, 16 elements each
  const int srow = t >> 1;
  const int skk  = (t & 1) * 16;
  const float*  Lrow = L  + (((size_t)b * Nn + row0 + srow) * Nn + skk);
  const __bf16* Trow = Tt + (((size_t)b * Dn + srow) * Nn + skk);
  __bf16* const apA[2] = { &Ash[0][srow * 36 + skk], &Ash[1][srow * 36 + skk] };
  const unsigned tdst[2] = {
    (unsigned)(uintptr_t)&Tsh[0][srow * 40 + skk],   // addr[31:0] = LDS offset
    (unsigned)(uintptr_t)&Tsh[1][srow * 40 + skk] };

  f4 r[4];                                 // staged L regs (next chunk)
  #define LOAD_L(K0) { _Pragma("unroll") \
      for (int i = 0; i < 4; ++i) r[i] = ((const f4*)(Lrow + (K0)))[i]; }
  #define ISSUE_TT(K0, BUF) { \
      async_b128(tdst[BUF],      Trow + (K0));     \
      async_b128(tdst[BUF] + 16, Trow + (K0) + 8); }

  // prologue: chunk 0 in flight
  ISSUE_TT(0, 0);
  LOAD_L(0);

  for (int kc = 0; kc < Nn / 32; ++kc) {
    const int cur = kc & 1;
    const int nxt = cur ^ 1;
    // commit staged L regs for chunk kc (cvt f32->bf16, 8B ds stores)
    {
      __bf16* p = apA[cur];
      #pragma unroll
      for (int i = 0; i < 4; ++i) ((v4bf*)p)[i] = cvt4(r[i]);
    }
    // prefetch chunk kc+1
    if (kc + 1 < Nn / 32) {
      ISSUE_TT((kc + 1) * 32, nxt);
      LOAD_L((kc + 1) * 32);
      // chunk kc's async pair retired; the 2 just issued may stay outstanding
      asm volatile("s_wait_asynccnt 2" ::: "memory");
    } else {
      asm volatile("s_wait_asynccnt 0" ::: "memory");
    }
    __syncthreads();   // implied s_wait_dscnt 0: buf[cur] fully populated

    // A fragment: 16x32 bf16, row m = 16w+ll, K = 16*hi + j (contiguous, same order as B)
    V16 a;
    {
      const __bf16* ap = &Ash[cur][(w * 16 + ll) * 36 + hi * 16];
      a.q[0] = *(const v4bf*)(ap);
      a.q[1] = *(const v4bf*)(ap + 4);
      a.q[2] = *(const v4bf*)(ap + 8);
      a.q[3] = *(const v4bf*)(ap + 12);
    }
    #pragma unroll
    for (int ct = 0; ct < 8; ++ct) {
      // B fragment: 32x16, col n = ll, K = 16*hi + j (Tsh column-major in n)
      V16 bb;
      const __bf16* bp = &Tsh[cur][(ct * 16 + ll) * 40 + hi * 16];
      bb.q[0] = *(const v4bf*)(bp);
      bb.q[1] = *(const v4bf*)(bp + 4);
      bb.q[2] = *(const v4bf*)(bp + 8);
      bb.q[3] = *(const v4bf*)(bp + 12);
      c[ct] = __builtin_amdgcn_wmma_f32_16x16x32_bf16(
          false, a.v, false, bb.v, (short)0, c[ct], false, false);
    }
  }
  #undef LOAD_L
  #undef ISSUE_TT

  // epilogue: tanh(msg) -> bf16, row-major (A-friendly for kernel 3)
  #pragma unroll
  for (int ct = 0; ct < 8; ++ct) {
    F8 u; u.v = c[ct];
    #pragma unroll
    for (int rr = 0; rr < 8; ++rr) {
      const int rowg = row0 + w * 16 + rr + hi * 8;  // C layout: lanes>=16 hold M+8
      const int col  = ct * 16 + ll;
      M[((size_t)b * Nn + rowg) * Dn + col] = (__bf16)tanhf(u.f[rr]);
    }
  }
}

// ---------------------------------------------------------------------------
// Kernel 3: state = tanh(x@W1 + tanh(msg)@W2); fused readout state@Wl summed
// over the block's 128 rows -> deterministic per-block partial sum.
// W1/W2 staged transposed in LDS once per block. grid = 512 blocks, 256 thr.
// ---------------------------------------------------------------------------
__global__ void __launch_bounds__(256)
k3_state_readout(const float* __restrict__ x, const __bf16* __restrict__ M,
                 const float* __restrict__ W1, const float* __restrict__ W2,
                 const float* __restrict__ Wl, float* __restrict__ partials) {
  __shared__ __bf16 W1s[Hn * 132];        // [h][d] transposed, stride 132 halfs
  __shared__ __bf16 W2s[Hn * 132];
  __shared__ float  Wls[Hn];
  __shared__ float  wsum[8];
  const int t    = threadIdx.x;
  const int b    = blockIdx.x >> 3;
  const int row0 = (blockIdx.x & 7) * 128;
  const int w    = t >> 5;
  const int lane = t & 31;
  const int ll   = lane & 15;
  const int hi   = lane >> 4;

  // stage W1,W2 transposed (one time, f32 -> bf16)
  {
    const int d  = t >> 1;
    const int h0 = (t & 1) * 32;
    const float* p1 = W1 + d * Hn + h0;
    const float* p2 = W2 + d * Hn + h0;
    #pragma unroll
    for (int i = 0; i < 32; ++i) {
      W1s[(h0 + i) * 132 + d] = (__bf16)p1[i];
      W2s[(h0 + i) * 132 + d] = (__bf16)p2[i];
    }
    if (t < Hn) Wls[t] = Wl[t];
  }
  __syncthreads();

  v8f c[4];
  const v8f zero = {0.f, 0.f, 0.f, 0.f, 0.f, 0.f, 0.f, 0.f};
  #pragma unroll
  for (int i = 0; i < 4; ++i) c[i] = zero;

  const int rowg = row0 + w * 16 + ll;    // per-lane A row
  const float*  xp = x + ((size_t)b * Nn + rowg) * Dn;
  const __bf16* mp = M + ((size_t)b * Nn + rowg) * Dn;

  #pragma unroll
  for (int k0 = 0; k0 < Dn; k0 += 32) {
    const int kb = k0 + hi * 8;
    V16 a1, a2;
    // a1: x (f32 -> bf16 on the fly), K halves [kb..kb+7] and [kb+16..kb+23]
    a1.q[0] = cvt4(*(const f4*)(xp + kb));
    a1.q[1] = cvt4(*(const f4*)(xp + kb + 4));
    a1.q[2] = cvt4(*(const f4*)(xp + kb + 16));
    a1.q[3] = cvt4(*(const f4*)(xp + kb + 20));
    // a2: tanh(msg) already bf16, 16B-aligned vector loads
    { V8 u0, u1;
      u0.v = *(const v8bf*)(mp + kb);
      u1.v = *(const v8bf*)(mp + kb + 16);
      a2.q[0] = u0.q[0]; a2.q[1] = u0.q[1];
      a2.q[2] = u1.q[0]; a2.q[3] = u1.q[1]; }

    #pragma unroll
    for (int ct = 0; ct < 4; ++ct) {
      const __bf16* b1p = &W1s[(ct * 16 + ll) * 132 + k0 + hi * 8];
      const __bf16* b2p = &W2s[(ct * 16 + ll) * 132 + k0 + hi * 8];
      V16 b1, b2;
      b1.q[0] = *(const v4bf*)(b1p);     b1.q[1] = *(const v4bf*)(b1p + 4);
      b1.q[2] = *(const v4bf*)(b1p + 16); b1.q[3] = *(const v4bf*)(b1p + 20);
      b2.q[0] = *(const v4bf*)(b2p);     b2.q[1] = *(const v4bf*)(b2p + 4);
      b2.q[2] = *(const v4bf*)(b2p + 16); b2.q[3] = *(const v4bf*)(b2p + 20);
      c[ct] = __builtin_amdgcn_wmma_f32_16x16x32_bf16(
          false, a1.v, false, b1.v, (short)0, c[ct], false, false);
      c[ct] = __builtin_amdgcn_wmma_f32_16x16x32_bf16(
          false, a2.v, false, b2.v, (short)0, c[ct], false, false);
    }
  }

  // fused readout: sum over all rows/cols of tanh(state)*Wl[h]
  float acc = 0.f;
  #pragma unroll
  for (int ct = 0; ct < 4; ++ct) {
    F8 u; u.v = c[ct];
    const float wl = Wls[ct * 16 + ll];
    #pragma unroll
    for (int rr = 0; rr < 8; ++rr) acc += tanhf(u.f[rr]) * wl;
  }
  #pragma unroll
  for (int off = 16; off > 0; off >>= 1) acc += __shfl_xor(acc, off, 32);
  if (lane == 0) wsum[w] = acc;
  __syncthreads();
  if (t == 0) {
    float s = 0.f;
    #pragma unroll
    for (int i = 0; i < 8; ++i) s += wsum[i];
    partials[blockIdx.x] = s;             // deterministic, no atomics
  }
}

// ---------------------------------------------------------------------------
// Kernel 4: loss = mean_b log1p(exp(-label[b] * sum(partials[b])/(N*H)))
// ---------------------------------------------------------------------------
__global__ void k4_loss(const float* __restrict__ partials,
                        const float* __restrict__ label,
                        float* __restrict__ out) {
  const int l = threadIdx.x;
  float s = 0.f;
  for (int b = l; b < Bn; b += 32) {
    float sb = 0.f;
    #pragma unroll
    for (int j = 0; j < 8; ++j) sb += partials[b * 8 + j];
    const float logit = sb * (1.f / ((float)Nn * (float)Hn));
    s += log1pf(expf(-logit * label[b]));
  }
  #pragma unroll
  for (int off = 16; off > 0; off >>= 1) s += __shfl_xor(s, off, 32);
  if (l == 0) out[0] = s * (1.f / (float)Bn);
}

// ---------------------------------------------------------------------------
extern "C" void kernel_launch(void* const* d_in, const int* in_sizes, int n_in,
                              void* d_out, int out_size, void* d_ws, size_t ws_size,
                              hipStream_t stream) {
  const float* x   = (const float*)d_in[0];  // node_feat [B,N,D]
  const float* L   = (const float*)d_in[1];  // Laplacian [B,N,N]
  const float* lab = (const float*)d_in[2];  // label     [B]
  const float* W1  = (const float*)d_in[3];  // [D,H]
  const float* W2  = (const float*)d_in[4];  // [D,H]
  const float* Wl  = (const float*)d_in[5];  // [H,1]
  float* out = (float*)d_out;

  char* ws = (char*)d_ws;
  const size_t tdSz = (size_t)Bn * Dn * Nn * 2;      // 16 MB: Tt bf16 [B][D][N]
  __bf16* Tt       = (__bf16*)ws;
  __bf16* Mm       = (__bf16*)(ws + tdSz);           // 16 MB: tanh(msg) bf16 [B][N][D]
  float*  partials = (float*)(ws + 2 * tdSz);        // 512 floats

  k1_tanh_transpose<<<Bn * (Nn / 64) * (Dn / 64), 256, 0, stream>>>(x, Tt);
  k2_msg_gemm      <<<Bn * (Nn / 128),            256, 0, stream>>>(L, Tt, Mm);
  k3_state_readout <<<Bn * (Nn / 128),            256, 0, stream>>>(x, Mm, W1, W2, Wl, partials);
  k4_loss          <<<1, 32,                           0, stream>>>(partials, lab, out);
}
